// SCOFFDynamics_88175678587132
// MI455X (gfx1250) — matrix operand order repaired and checked by reading, not compiled, not run on hardware
//
#include <hip/hip_runtime.h>
#include <math.h>

// ---------------------------------------------------------------------------
// SCOFF dynamics fused kernel for MI455X (gfx1250, wave32, WMMA bf16).
//
// Precompute (pack kernels):
//   - w_ih / w_hh repacked to bf16 B fragments, k = 16*(lane/16) + e within each lane.
//   - Wlog[r,h] = (1/sqrt(128)) * sum_k keys[r,k] * wq[k,h]  (fuses query+logits)
//
// Main kernel: 16-row tiles (M = N*S = 32768 rows -> 2048 tiles), 8 waves/WG,
// one rule per wave, v_wmma_f32_16x16x32_bf16 everywhere. Weight fragments are
// prefetched one hb-iteration ahead (global_prefetch_b8) to overlap the
// GL2 -> WGP fill with the current iteration's 24 WMMAs.
// ---------------------------------------------------------------------------

typedef __bf16 v16bf __attribute__((ext_vector_type(16)));
typedef float  v8f   __attribute__((ext_vector_type(8)));

constexpr int kH      = 128;     // hidden
constexpr int kS      = 8;       // seq
constexpr int kN      = 4096;    // batch
constexpr int kR      = 8;       // rules
constexpr int kK      = 512;     // NHEAD*KEY_SIZE
constexpr int kG3     = 384;     // 3*H
constexpr int kNT     = 24;      // n-tiles of 16 per rule (3H/16)
constexpr int kKT     = 4;       // k-steps of 32 (H/32)
constexpr int kFragHalfs = 512;  // 32 lanes * 16 halves per B fragment

static __device__ __forceinline__ float sigm(float x) {
  return 1.0f / (1.0f + __expf(-x));
}

// Deterministic counter-hash uniform in (0,1) — stand-in for the JAX threefry
// stream (same value for a given logical index every call).
static __device__ __forceinline__ float hash_u01(unsigned x) {
  x *= 0x9E3779B9u;
  x ^= x >> 16; x *= 0x7FEB352Du;
  x ^= x >> 15; x *= 0x846CA68Bu;
  x ^= x >> 16;
  float u = (float)(x >> 8) * (1.0f / 16777216.0f);
  return fminf(fmaxf(u, 1e-10f), 0.9999999f);
}

static __device__ __forceinline__ v8f wmma_bf16(v16bf a, v16bf b, v8f c) {
  return __builtin_amdgcn_wmma_f32_16x16x32_bf16(
      /*neg_a=*/false, a, /*neg_b=*/false, b,
      /*c_mod=*/(short)0, c, /*reuse_a=*/false, /*reuse_b=*/false);
}

// ---------------------------------------------------------------------------
// Pack w_ih / w_hh into WMMA B fragments (bf16), contiguous k per lane half.
// Fragment (r, nt, kt): halves laid out [lane][e] with
//   n_local = lane&15,  k_local = (lane>=16 ? 16 : 0) + e
//   B[k_local, n_local] = w[r, nt*16 + n_local, kt*32 + k_local]
// ---------------------------------------------------------------------------
__global__ __launch_bounds__(256) void pack_gru_weights(
    const float* __restrict__ w_ih, const float* __restrict__ w_hh,
    __bf16* __restrict__ bih, __bf16* __restrict__ bhh) {
  const int per = kR * kNT * kKT * kFragHalfs;  // 393216
  int t = blockIdx.x * blockDim.x + threadIdx.x;
  int sel = t / per;
  int within = t - sel * per;
  int e = within & 15;
  int lane = (within >> 4) & 31;
  int fragIdx = within >> 9;
  int kt = fragIdx & 3;
  int q = fragIdx >> 2;
  int nt = q % kNT;
  int r = q / kNT;
  int g = nt * 16 + (lane & 15);
  int kl = ((lane >> 4) << 4) + e;
  int hcol = kt * 32 + kl;
  const float* src = sel ? w_hh : w_ih;
  float v = src[(r * kG3 + g) * kH + hcol];
  (sel ? bhh : bih)[within] = (__bf16)v;
}

// ---------------------------------------------------------------------------
// Wlog[r,h] = (1/sqrt(128)) * sum_k keys[r,k]*wq[k,h], packed as one B-fragment
// set [kt][lane][e] (rules 8..15 zero-padded).
// ---------------------------------------------------------------------------
__global__ __launch_bounds__(256) void pack_wlog(
    const float* __restrict__ keys, const float* __restrict__ wq,
    __bf16* __restrict__ blog) {
  int t = blockIdx.x * blockDim.x + threadIdx.x;  // 0..2047
  int e = t & 15;
  int lane = (t >> 4) & 31;
  int kt = t >> 9;
  int nl = lane & 15;
  int kl = ((lane >> 4) << 4) + e;
  int hcol = kt * 32 + kl;
  float acc = 0.0f;
  if (nl < kR) {
    for (int k = 0; k < kK; ++k) acc += keys[nl * kK + k] * wq[k * kH + hcol];
    acc *= 0.08838834764831845f;  // 1/sqrt(128)
  }
  blog[t] = (__bf16)acc;
}

// ---------------------------------------------------------------------------
// Main fused kernel: 16-row tile per workgroup, 8 waves (one rule each).
// ---------------------------------------------------------------------------
__global__ __launch_bounds__(256) void scoff_main(
    const float* __restrict__ z, const float* __restrict__ h,
    const float* __restrict__ b_ih, const float* __restrict__ b_hh,
    const v16bf* __restrict__ Bih, const v16bf* __restrict__ Bhh,
    const v16bf* __restrict__ Blog, float* __restrict__ out) {
  __shared__ __bf16 xs[16 * kH];      // z tile, bf16
  __shared__ __bf16 hs[16 * kH];      // h tile, bf16
  __shared__ float  hf[16 * kH];      // h tile, fp32 (for (1-z)*n + z*h)
  __shared__ float  outacc[16 * kH];  // fp32 accumulation over rules
  __shared__ float  attn_s[2][kR][kS];

  const int tile = blockIdx.x;
  const int tid = threadIdx.x;
  const int lane = tid & 31;
  const int wid = tid >> 5;  // wave id == rule id
  const size_t rowbase = (size_t)tile * 16 * kH;

  // ---- Stage tile: 2048 floats each of z and h (512 float4 per tensor) ----
  {
    const float4* zz = (const float4*)(z + rowbase);
    const float4* hh = (const float4*)(h + rowbase);
#pragma unroll
    for (int i = 0; i < 2; ++i) {
      int idx = tid + i * 256;
      float4 a = zz[idx];
      float4 b = hh[idx];
      int base = idx * 4;
      xs[base + 0] = (__bf16)a.x; xs[base + 1] = (__bf16)a.y;
      xs[base + 2] = (__bf16)a.z; xs[base + 3] = (__bf16)a.w;
      hs[base + 0] = (__bf16)b.x; hs[base + 1] = (__bf16)b.y;
      hs[base + 2] = (__bf16)b.z; hs[base + 3] = (__bf16)b.w;
      hf[base + 0] = b.x; hf[base + 1] = b.y;
      hf[base + 2] = b.z; hf[base + 3] = b.w;
      outacc[base + 0] = 0.0f; outacc[base + 1] = 0.0f;
      outacc[base + 2] = 0.0f; outacc[base + 3] = 0.0f;
    }
  }

  // ---- Warm the WGP for this wave's weight fragments (hb = 0) while the
  //      tile staging / barrier is in flight (lowers to global_prefetch_b8).
  {
    const int r0 = wid;
    const int f_r = (r0 * kNT + 0) * 4;        // gate r, hb=0
    const int f_z = (r0 * kNT + 8) * 4;        // gate z, hb=0
    const int f_n = (r0 * kNT + 16) * 4;       // gate n, hb=0
#pragma unroll
    for (int kt = 0; kt < 4; ++kt) {
      __builtin_prefetch(&Bih[(f_r + kt) * 32 + lane], 0, 3);
      __builtin_prefetch(&Bih[(f_z + kt) * 32 + lane], 0, 3);
      __builtin_prefetch(&Bih[(f_n + kt) * 32 + lane], 0, 3);
      __builtin_prefetch(&Bhh[(f_r + kt) * 32 + lane], 0, 3);
      __builtin_prefetch(&Bhh[(f_z + kt) * 32 + lane], 0, 3);
      __builtin_prefetch(&Bhh[(f_n + kt) * 32 + lane], 0, 3);
    }
  }
  __syncthreads();

  // ---- Build A fragments (16x32 bf16, k = 16*hi + element, same order as B) ----
  const int m = lane & 15;       // row within tile for A/B lanes
  const int hi = lane >> 4;      // lane half
  v16bf ax[4], ah[4];
#pragma unroll
  for (int kt = 0; kt < 4; ++kt) {
#pragma unroll
    for (int j = 0; j < 8; ++j) {
      int kb = 2 * j + hi * 16;
      int k = kt * 32 + kb;
      ax[kt][2 * j + 0] = xs[m * kH + k];
      ax[kt][2 * j + 1] = xs[m * kH + k + 1];
      ah[kt][2 * j + 0] = hs[m * kH + k];
      ah[kt][2 * j + 1] = hs[m * kH + k + 1];
    }
  }

  // ---- Wave 0: logits WMMA + gumbel softmax + straight-through hard ----
  // D layout: element v of lane L is (row = v + 8*(L>=16), col = L&15).
  // row = ng*8 + s, col = rule  =>  each lane owns all 8 s for one (ng, rule).
  if (wid == 0) {
    v8f lacc = {0.f, 0.f, 0.f, 0.f, 0.f, 0.f, 0.f, 0.f};
#pragma unroll
    for (int kt = 0; kt < 4; ++kt)
      lacc = wmma_bf16(ah[kt], Blog[kt * 32 + lane], lacc);
    int rr = lane & 15;
    int ng = hi;
    if (rr < kR) {
      int nglob = tile * 2 + ng;
      float t8[8];
      float mx = -1e30f;
#pragma unroll
      for (int s8 = 0; s8 < 8; ++s8) {
        unsigned idx = (unsigned)((nglob * kR + rr) * kS + s8);
        float u = hash_u01(idx + 0x42424242u);
        float gmb = -__logf(-__logf(u));
        float v = lacc[s8] + gmb;  // TAU == 1.0
        t8[s8] = v;
        mx = fmaxf(mx, v);
      }
      float den = 0.0f;
#pragma unroll
      for (int s8 = 0; s8 < 8; ++s8) { t8[s8] = __expf(t8[s8] - mx); den += t8[s8]; }
      float inv = 1.0f / den;
      int amax = 0;
      float best = -1.0f;
#pragma unroll
      for (int s8 = 0; s8 < 8; ++s8) {
        float ys = t8[s8] * inv;
        t8[s8] = ys;
        if (ys > best) { best = ys; amax = s8; }
      }
#pragma unroll
      for (int s8 = 0; s8 < 8; ++s8) {
        float yh = (s8 == amax) ? 1.0f : 0.0f;
        attn_s[ng][rr][s8] = (yh - t8[s8]) + t8[s8];  // straight-through value
      }
    }
  }
  __syncthreads();

  // ---- GRU-pool for this wave's rule, weighted into outacc ----
  const int r = wid;
  float aw[8];
#pragma unroll
  for (int s8 = 0; s8 < 8; ++s8) aw[s8] = attn_s[hi][r][s8];  // s == v index

  for (int hb = 0; hb < 8; ++hb) {
    // Prefetch next iteration's 6 fragments (24 KB/rule total working set,
    // L2-resident): overlap GL2 -> WGP fill with this iteration's 24 WMMAs.
    if (hb < 7) {
      const int p_r = (r * kNT + (0 * 8 + hb + 1)) * 4;
      const int p_z = (r * kNT + (1 * 8 + hb + 1)) * 4;
      const int p_n = (r * kNT + (2 * 8 + hb + 1)) * 4;
      __builtin_prefetch(&Bih[p_r * 32 + lane], 0, 3);
      __builtin_prefetch(&Bih[p_z * 32 + lane], 0, 3);
      __builtin_prefetch(&Bih[p_n * 32 + lane], 0, 3);
      __builtin_prefetch(&Bhh[p_r * 32 + lane], 0, 3);
      __builtin_prefetch(&Bhh[p_z * 32 + lane], 0, 3);
      __builtin_prefetch(&Bhh[p_n * 32 + lane], 0, 3);
    }

    v8f a_ir = {0.f,0.f,0.f,0.f,0.f,0.f,0.f,0.f};
    v8f a_iz = {0.f,0.f,0.f,0.f,0.f,0.f,0.f,0.f};
    v8f a_in = {0.f,0.f,0.f,0.f,0.f,0.f,0.f,0.f};
    v8f a_hr = {0.f,0.f,0.f,0.f,0.f,0.f,0.f,0.f};
    v8f a_hz = {0.f,0.f,0.f,0.f,0.f,0.f,0.f,0.f};
    v8f a_hn = {0.f,0.f,0.f,0.f,0.f,0.f,0.f,0.f};
    const int fb_r = (r * kNT + (0 * 8 + hb)) * 4;
    const int fb_z = (r * kNT + (1 * 8 + hb)) * 4;
    const int fb_n = (r * kNT + (2 * 8 + hb)) * 4;
#pragma unroll
    for (int kt = 0; kt < 4; ++kt) {
      v16bf bir_f = Bih[(fb_r + kt) * 32 + lane];
      v16bf biz_f = Bih[(fb_z + kt) * 32 + lane];
      v16bf bin_f = Bih[(fb_n + kt) * 32 + lane];
      v16bf bhr_f = Bhh[(fb_r + kt) * 32 + lane];
      v16bf bhz_f = Bhh[(fb_z + kt) * 32 + lane];
      v16bf bhn_f = Bhh[(fb_n + kt) * 32 + lane];
      a_ir = wmma_bf16(ax[kt], bir_f, a_ir);
      a_iz = wmma_bf16(ax[kt], biz_f, a_iz);
      a_in = wmma_bf16(ax[kt], bin_f, a_in);
      a_hr = wmma_bf16(ah[kt], bhr_f, a_hr);
      a_hz = wmma_bf16(ah[kt], bhz_f, a_hz);
      a_hn = wmma_bf16(ah[kt], bhn_f, a_hn);
    }
    const int ch = hb * 16 + m;
    const float bi_r = b_ih[r * kG3 + ch];
    const float bi_z = b_ih[r * kG3 + kH + ch];
    const float bi_n = b_ih[r * kG3 + 2 * kH + ch];
    const float bh_r = b_hh[r * kG3 + ch];
    const float bh_z = b_hh[r * kG3 + kH + ch];
    const float bh_n = b_hh[r * kG3 + 2 * kH + ch];
#pragma unroll
    for (int v = 0; v < 8; ++v) {
      int mrow = v + hi * 8;
      float rg = sigm(a_ir[v] + bi_r + a_hr[v] + bh_r);
      float zg = sigm(a_iz[v] + bi_z + a_hz[v] + bh_z);
      float ngt = tanhf(a_in[v] + bi_n + rg * (a_hn[v] + bh_n));
      float hp = hf[mrow * kH + ch];
      float hn = (1.0f - zg) * ngt + zg * hp;
      atomicAdd(&outacc[mrow * kH + ch], aw[v] * hn);  // ds_add_f32
    }
  }
  __syncthreads();

  // ---- Write out 16x128 fp32 tile ----
  {
    float4* og = (float4*)(out + rowbase);
    const float4* oa = (const float4*)outacc;
#pragma unroll
    for (int i = 0; i < 2; ++i) og[tid + i * 256] = oa[tid + i * 256];
  }
}

// ---------------------------------------------------------------------------
extern "C" void kernel_launch(void* const* d_in, const int* in_sizes, int n_in,
                              void* d_out, int out_size, void* d_ws, size_t ws_size,
                              hipStream_t stream) {
  (void)in_sizes; (void)n_in; (void)out_size; (void)ws_size;
  const float* z    = (const float*)d_in[0];
  const float* h    = (const float*)d_in[1];
  const float* w_ih = (const float*)d_in[2];
  const float* w_hh = (const float*)d_in[3];
  const float* b_ih = (const float*)d_in[4];
  const float* b_hh = (const float*)d_in[5];
  const float* keys = (const float*)d_in[6];
  const float* wq   = (const float*)d_in[7];
  float* out = (float*)d_out;

  const size_t per = (size_t)kR * kNT * kKT * kFragHalfs;  // 393216 bf16
  __bf16* bih  = (__bf16*)d_ws;
  __bf16* bhh  = bih + per;
  __bf16* blog = bhh + per;  // 2048 bf16

  pack_gru_weights<<<(2 * (int)per) / 256, 256, 0, stream>>>(w_ih, w_hh, bih, bhh);
  pack_wlog<<<2048 / 256, 256, 0, stream>>>(keys, wq, blog);
  scoff_main<<<(kN * kS) / 16, 256, 0, stream>>>(
      z, h, b_ih, b_hh, (const v16bf*)bih, (const v16bf*)bhh,
      (const v16bf*)blog, out);
}
